// DotProductAttention_74990128988197
// MI455X (gfx1250) — matrix operand order repaired and checked by reading, not compiled, not run on hardware
//
#include <hip/hip_runtime.h>

typedef __attribute__((ext_vector_type(16))) _Float16 v16h;
typedef __attribute__((ext_vector_type(8)))  _Float16 v8h;
typedef __attribute__((ext_vector_type(8)))  float    v8f;

#define SQ 2048
#define BB 2
#define NP 16
#define HN 64
#define SROW (BB*NP*HN)      // floats between consecutive seq rows = 2048
#define BLOCK_M 128          // query rows per workgroup (8 waves x 16 rows)
#define KT 64                // keys processed per main-loop iteration

static_assert(SQ % BLOCK_M == 0 && SQ % KT == 0, "tiling");

__device__ __forceinline__ v8f wmma16(v16h a, v16h b, v8f c) {
  // D = A(16x32 f16) * B(32x16 f16) + C(16x16 f32)
  return __builtin_amdgcn_wmma_f32_16x16x32_f16(
      /*neg_a=*/false, a, /*neg_b=*/false, b,
      /*c_mod=*/(short)0, c, /*reuse_a=*/false, /*reuse_b=*/false);
}

// Exchange value with the other 16-lane half (lane ^ 16), as a VALU lane op.
__device__ __forceinline__ float xswap16(float v) {
#if __has_builtin(__builtin_amdgcn_permlanex16)
  int i = __float_as_int(v);
  i = __builtin_amdgcn_permlanex16(i, i, 0x76543210, (int)0xFEDCBA98, false, false);
  return __int_as_float(i);
#else
  return __shfl_xor(v, 16, 32);
#endif
}

__device__ __forceinline__ v16h packA(v8h lo, v8h hi) {
  v16h a;
  #pragma unroll
  for (int i = 0; i < 8; ++i) { a[i] = lo[i]; a[8 + i] = hi[i]; }
  return a;
}

__global__ __launch_bounds__(256)
void fa_fwd_kernel(const float* __restrict__ Q, const float* __restrict__ K,
                   const float* __restrict__ V, float* __restrict__ O) {
  // double-buffered K / V^T tiles: 2 x (8KB + 8KB) = 32KB LDS
  __shared__ alignas(32) _Float16 sK [2 * KT * HN];   // [buf][key][h]
  __shared__ alignas(32) _Float16 sVt[2 * HN * KT];   // [buf][h][key]

  const int tid  = threadIdx.x;
  const int lane = tid & 31;
  const int wv   = tid >> 5;       // wave 0..7
  const int half = lane >> 4;      // lane-half
  const int l16  = lane & 15;      // this lane's query column

  const int head = blockIdx.y;     // b*NP + n
  const int b = head >> 4;
  const int n = head & 15;
  const size_t headoff = (size_t)(b * NP + n) * HN;

  const float* __restrict__ Qh = Q + headoff;
  const float* __restrict__ Kh = K + headoff;
  const float* __restrict__ Vh = V + headoff;
  float*       __restrict__ Oh = O + headoff;

  // ---- cooperative fill of buffer `bi` with key tile starting at kb2 ----
  auto fill = [&](int bi, int kb2) {
    _Float16* dK = sK  + bi * (KT * HN);
    _Float16* dV = sVt + bi * (HN * KT);
    #pragma unroll
    for (int i = 0; i < 8; ++i) {
      const int pid = i * 256 + tid;       // 0..2047 (float2 pairs)
      const int kn  = pid >> 5;            // key row 0..63
      const int h2  = (pid & 31) << 1;     // even h
      const float2 kv = *(const float2*)(Kh + (size_t)(kb2 + kn) * SROW + h2);
      const float2 vv = *(const float2*)(Vh + (size_t)(kb2 + kn) * SROW + h2);
      dK[kn * HN + h2]       = (_Float16)kv.x;   // packs to v_cvt_pk + b32 store
      dK[kn * HN + h2 + 1]   = (_Float16)kv.y;
      dV[(h2    ) * KT + kn] = (_Float16)vv.x;   // transposed scatter
      dV[(h2 + 1) * KT + kn] = (_Float16)vv.y;
    }
    if (kb2 + KT < SQ) {   // prefetch tile two ahead (global_prefetch_b8)
      const float* pb = (tid < 128) ? Kh : Vh;
      const int t2 = tid & 127;
      __builtin_prefetch(pb + (size_t)(kb2 + KT + (t2 >> 1)) * SROW + (t2 & 1) * 32, 0, 0);
    }
  };

  // ---- Q as B-operand for S^T = K * Q^T : fully contiguous loads ----
  // B layout (16-bit, 32x16): col N = lane&15 (query), element e: k = (e/8)*16 + half*8 + e%8
  const int qrow = blockIdx.x * BLOCK_M + wv * 16 + l16;
  const float* __restrict__ Qp = Qh + (size_t)qrow * SROW;
  v16h bQ0, bQ1;
  #pragma unroll
  for (int e = 0; e < 16; ++e) {
    bQ0[e] = (_Float16)Qp[      (e >> 3) * 16 + half * 8 + (e & 7)];
    bQ1[e] = (_Float16)Qp[32 + (e >> 3) * 16 + half * 8 + (e & 7)];
  }

  // context^T accumulators: 64(h) x 16(q) in 4 C/D tiles (row=h, col=query)
  v8f ct[4];
  #pragma unroll
  for (int t = 0; t < 4; ++t) ct[t] = (v8f){};
  float m_i = -3.0e38f, l_i = 0.0f;   // per-query softmax state (query = l16)

  fill(0, 0);
  __syncthreads();

  int buf = 0;
  for (int kb = 0; kb < SQ; kb += KT) {
    // ---- kick off next tile's fill into the other buffer (overlaps compute)
    if (kb + KT < SQ) fill(buf ^ 1, kb + KT);

    const _Float16* cK = sK  + buf * (KT * HN);
    const _Float16* cV = sVt + buf * (HN * KT);

    // ---- S^T = K * Q^T : 4 key-subtiles x 2 k-steps ----
    // A layout (16-bit 16x32): row M = lane&15 (key), element e:
    // k = (e/8)*16 + half*8 + e%8  -> two contiguous v8h LDS reads.
    v8f st[4];
    #pragma unroll
    for (int j = 0; j < 4; ++j) {
      const _Float16* krow = &cK[(j * 16 + l16) * HN];
      v8f acc = (v8f){};
      v16h a0 = packA(*(const v8h*)&krow[     half * 8],
                      *(const v8h*)&krow[16 + half * 8]);
      acc = wmma16(a0, bQ0, acc);
      v16h a1 = packA(*(const v8h*)&krow[32 + half * 8],
                      *(const v8h*)&krow[48 + half * 8]);
      acc = wmma16(a1, bQ1, acc);
      st[j] = acc;   // row = key (r + half_src*8), col = query (l16)
    }

    // ---- online softmax: each lane holds 32 of this tile's 64 keys for
    //      query l16; the other 32 live in lane^16. ----
    float mx = -3.0e38f;
    #pragma unroll
    for (int j = 0; j < 4; ++j)
      #pragma unroll
      for (int r = 0; r < 8; ++r) mx = fmaxf(mx, st[j][r]);
    mx = fmaxf(mx, xswap16(mx));
    const float mnew = fmaxf(m_i, mx);
    const float corr = __expf(m_i - mnew);
    m_i = mnew;
    float rs = 0.0f;
    #pragma unroll
    for (int j = 0; j < 4; ++j)
      #pragma unroll
      for (int r = 0; r < 8; ++r) {
        const float p = __expf(st[j][r] - mnew);
        st[j][r] = p;
        rs += p;
      }
    rs += xswap16(rs);
    l_i = l_i * corr + rs;
    #pragma unroll
    for (int t = 0; t < 4; ++t)
      #pragma unroll
      for (int r = 0; r < 8; ++r) ct[t][r] *= corr;

    // ---- context^T += V^T * P^T ----
    // P^T B-operand built in registers: element e needs key s*32 + half*16 + e,
    // sourced from subtile (2s+half); lane-half crossing via permlanex16.
    #pragma unroll
    for (int s = 0; s < 2; ++s) {
      v16h bp;
      #pragma unroll
      for (int r = 0; r < 8; ++r) {
        const float olo = st[2 * s][r];
        const float ohi = st[2 * s + 1][r];
        const float xlo = xswap16(olo);
        const float xhi = xswap16(ohi);
        bp[r]     = (_Float16)(half ? xhi : olo);
        bp[r + 8] = (_Float16)(half ? ohi : xlo);
      }
      #pragma unroll
      for (int t = 0; t < 4; ++t) {
        const _Float16* vrow = &cV[(t * 16 + l16) * KT + s * 32];
        v16h aV = packA(*(const v8h*)&vrow[half * 16],
                        *(const v8h*)&vrow[half * 16 + 8]);
        ct[t] = wmma16(aV, bp, ct[t]);
      }
    }

    // single barrier per iteration: all waves done reading `buf` and done
    // writing `buf^1` before anyone proceeds (next iter overwrites `buf`).
    __syncthreads();
    buf ^= 1;
  }

  // ---- epilogue: normalize; per-lane contiguous b128 stores ----
  const float inv = 1.0f / l_i;
  float* op = Oh + (size_t)qrow * SROW;   // this lane's query row
  #pragma unroll
  for (int t = 0; t < 4; ++t) {
    // h = t*16 + half*8 + r  (r = 0..7) -> contiguous 8 floats
    float4 lo4 = make_float4(ct[t][0] * inv, ct[t][1] * inv,
                             ct[t][2] * inv, ct[t][3] * inv);
    float4 hi4 = make_float4(ct[t][4] * inv, ct[t][5] * inv,
                             ct[t][6] * inv, ct[t][7] * inv);
    *(float4*)(op + t * 16 + half * 8)     = lo4;
    *(float4*)(op + t * 16 + half * 8 + 4) = hi4;
  }
}

extern "C" void kernel_launch(void* const* d_in, const int* in_sizes, int n_in,
                              void* d_out, int out_size, void* d_ws, size_t ws_size,
                              hipStream_t stream) {
  (void)in_sizes; (void)n_in; (void)out_size; (void)d_ws; (void)ws_size;
  const float* Q = (const float*)d_in[0];
  const float* K = (const float*)d_in[1];
  const float* V = (const float*)d_in[2];
  float*       O = (float*)d_out;
  dim3 grid(SQ / BLOCK_M, BB * NP);
  dim3 block(256);
  fa_fwd_kernel<<<grid, block, 0, stream>>>(Q, K, V, O);
}
